// CrossAttention_53042846106091
// MI455X (gfx1250) — matrix operand order repaired and checked by reading, not compiled, not run on hardware
//
#include <hip/hip_runtime.h>
#include <hip/hip_bf16.h>

// ---------------------------------------------------------------------------
// CDNA5 (gfx1250) cross-attention: bf16 WMMA everywhere, f32 accumulation.
// ---------------------------------------------------------------------------

typedef __attribute__((ext_vector_type(16))) __bf16 v16bf;
typedef __attribute__((ext_vector_type(8)))  __bf16 v8bf;
typedef __attribute__((ext_vector_type(8)))  float  v8f;
typedef __attribute__((ext_vector_type(4)))  float  v4f;

// Problem constants (fixed by the reference)
constexpr int BATCH = 2;
constexpr int HEADS = 16;
constexpr int DH    = 64;
constexpr int TSEQ  = 2048;
constexpr int EMB   = 1024;
constexpr int MROWS = BATCH * TSEQ;   // 4096

// --- fragment builders ------------------------------------------------------
// A fragment, 16x32 bf16 (ISA 7.12.2): lane L<16 -> row M=L, K = {0..7,16..23};
// lane L>=16 -> row M=L-16, K = {8..15,24..31}. Caller passes p = row + kk + half*8.
__device__ __forceinline__ v16bf afrag_f32(const float* __restrict__ p) {
  v4f f0 = *(const v4f*)(p);      v4f f1 = *(const v4f*)(p + 4);
  v4f f2 = *(const v4f*)(p + 16); v4f f3 = *(const v4f*)(p + 20);
  v16bf a;
#pragma unroll
  for (int i = 0; i < 4; ++i) {
    a[i]      = (__bf16)f0[i];
    a[4 + i]  = (__bf16)f1[i];
    a[8 + i]  = (__bf16)f2[i];
    a[12 + i] = (__bf16)f3[i];
  }
  return a;
}
// B fragment, 32x16 bf16: lane L -> col N=L%16, K = half*8 + {0..7,16..23}.
__device__ __forceinline__ v16bf bfrag_f32(const float* __restrict__ p) {
  v4f f0 = *(const v4f*)(p);      v4f f1 = *(const v4f*)(p + 4);
  v4f f2 = *(const v4f*)(p + 16); v4f f3 = *(const v4f*)(p + 20);
  v16bf b;
#pragma unroll
  for (int i = 0; i < 4; ++i) {
    b[i]      = (__bf16)f0[i];
    b[4 + i]  = (__bf16)f1[i];
    b[8 + i]  = (__bf16)f2[i];
    b[12 + i] = (__bf16)f3[i];
  }
  return b;
}
// Same fragments from bf16 memory (16B/32B-aligned contiguous runs).
__device__ __forceinline__ v16bf afrag_bf16(const __bf16* __restrict__ p) {
  v8bf lo = *(const v8bf*)(p);
  v8bf hi = *(const v8bf*)(p + 16);
  v16bf a;
#pragma unroll
  for (int i = 0; i < 8; ++i) { a[i] = lo[i]; a[8 + i] = hi[i]; }
  return a;
}
__device__ __forceinline__ v16bf bfrag_bf16(const __bf16* __restrict__ p) {
  return *(const v16bf*)(p);   // 16 contiguous bf16, 32B aligned
}

// ---------------------------------------------------------------------------
// GEMM: C[M,N] = (A[M,K] @ W[N,K]^T + bias[N]) * oscale
// MODE 0: store bf16 as [B,H,T,Dh]   (Q (oscale=1/8), K projections)
// MODE 1: store bf16 as [B,H,Dh,T]   (V projection, transposed for attention)
// MODE 2: store f32  as [M,N]        (output projection)
// block = 128 threads (4 wave32); wave computes a 32x64 strip (2x4 tiles,
// 8 WMMAs per K-step, B frags reused across both row tiles).
// grid = (M/32, N/256)
// ---------------------------------------------------------------------------
template <int MODE>
__global__ __launch_bounds__(128) void proj_gemm_wmma(
    const float* __restrict__ A, const float* __restrict__ W,
    const float* __restrict__ bias, void* __restrict__ Cout, int K,
    float oscale) {
  const int lane = threadIdx.x & 31;
  const int wave = threadIdx.x >> 5;
  const int half = lane >> 4;
  const int lm   = lane & 15;
  const int m0   = blockIdx.x * 32;
  const int n0   = (blockIdx.y * 4 + wave) * 64;

  v8f acc[2][4];
#pragma unroll
  for (int u = 0; u < 2; ++u)
#pragma unroll
    for (int t = 0; t < 4; ++t) acc[u][t] = v8f{};

  const float* arow0 = A + (size_t)(m0 + lm) * K;
  const float* arow1 = A + (size_t)(m0 + 16 + lm) * K;

  for (int kk = 0; kk < K; kk += 32) {
    __builtin_prefetch(arow0 + kk + 128, 0, 0);           // global_prefetch_b8
    const v16bf a0 = afrag_f32(arow0 + kk + half * 8);
    const v16bf a1 = afrag_f32(arow1 + kk + half * 8);
#pragma unroll
    for (int t = 0; t < 4; ++t) {
      const float* brow = W + (size_t)(n0 + t * 16 + lm) * K + kk + half * 8;
      const v16bf b = bfrag_f32(brow);
      acc[0][t] = __builtin_amdgcn_wmma_f32_16x16x32_bf16(
          false, a0, false, b, (short)0, acc[0][t], false, false);
      acc[1][t] = __builtin_amdgcn_wmma_f32_16x16x32_bf16(
          false, a1, false, b, (short)0, acc[1][t], false, false);
    }
  }

#pragma unroll
  for (int u = 0; u < 2; ++u) {
#pragma unroll
    for (int t = 0; t < 4; ++t) {
      const int n = n0 + t * 16 + lm;
      const float bv = bias[n];
#pragma unroll
      for (int r = 0; r < 8; ++r) {
        const int m = m0 + u * 16 + r + half * 8;   // C layout: M = r + half*8
        const float val = (acc[u][t][r] + bv) * oscale;
        if (MODE == 2) {
          ((float*)Cout)[(size_t)m * EMB + n] = val;
        } else {
          const int b  = m >> 11;                   // / TSEQ
          const int tq = m & (TSEQ - 1);
          const int h  = n >> 6;                    // / DH
          const int d  = n & (DH - 1);
          if (MODE == 0) {
            ((__bf16*)Cout)[(((size_t)(b * HEADS + h)) * TSEQ + tq) * DH + d] =
                (__bf16)val;
          } else {  // MODE 1: [B,H,Dh,T]
            ((__bf16*)Cout)[(((size_t)(b * HEADS + h)) * DH + d) * TSEQ + tq] =
                (__bf16)val;
          }
        }
      }
    }
  }
}

// ---------------------------------------------------------------------------
// Flash attention over one (b,h): Q[T,Dh] bf16 (pre-scaled by 1/sqrt(Dh)),
// K[T,Dh] bf16, Vt[Dh,T] bf16. Wave handles 16 query rows; the raw score
// tile is transposed through per-wave LDS FIRST, so softmax row statistics
// are per-lane scalars (one shfl_xor(16) per reduction instead of bpermute
// trees), and the exp'ed P lands directly in A-fragment registers.
// grid = (T/64, B*H), block = 128 (4 waves).
// ---------------------------------------------------------------------------
__global__ __launch_bounds__(128) void flash_attn_wmma(
    const __bf16* __restrict__ Qb, const __bf16* __restrict__ Kb,
    const __bf16* __restrict__ Vt, float* __restrict__ Attn) {
  __shared__ float lds_s[4][16][36];   // per-wave 16x32 S tile (f32, +4 pad)
  __shared__ float lds_a[4][16];       // per-wave row-factor broadcast line

  const int lane = threadIdx.x & 31;
  const int wave = threadIdx.x >> 5;
  const int half = lane >> 4;
  const int lm   = lane & 15;
  const int bh   = blockIdx.y;                       // b*HEADS + h
  const int q0   = blockIdx.x * 64 + wave * 16;

  // Q A-fragments for dh 0..31 and 32..63, resident for the whole loop.
  const __bf16* qrow = Qb + ((size_t)bh * TSEQ + q0 + lm) * DH;
  const v16bf qf0 = afrag_bf16(qrow + 0  + half * 8);
  const v16bf qf1 = afrag_bf16(qrow + 32 + half * 8);

  // Per-lane softmax state for row lm (both half-lanes carry the same row).
  float m_l = -1e30f, l_l = 0.f;
  v8f o[4] = {v8f{}, v8f{}, v8f{}, v8f{}};

  for (int kt = 0; kt < TSEQ; kt += 32) {
    // ---- scores: S[16x32] = Q(16x64) . K(32x64)^T, two 16x16 C tiles -------
    v8f s[2] = {v8f{}, v8f{}};
#pragma unroll
    for (int sub = 0; sub < 2; ++sub) {
      const __bf16* krow = Kb + ((size_t)bh * TSEQ + kt + sub * 16 + lm) * DH;
      const v16bf b0 = afrag_bf16(krow + 0  + half * 8);
      s[sub] = __builtin_amdgcn_wmma_f32_16x16x32_bf16(
          false, qf0, false, b0, (short)0, s[sub], false, false);
      const v16bf b1 = afrag_bf16(krow + 32 + half * 8);
      s[sub] = __builtin_amdgcn_wmma_f32_16x16x32_bf16(
          false, qf1, false, b1, (short)0, s[sub], false, false);
    }

    // ---- transpose S (C layout) -> per-lane rows via private LDS tile -----
    // Same-wave DS ops are in-order; no cross-wave sharing -> no barrier.
#pragma unroll
    for (int r = 0; r < 8; ++r) {
      lds_s[wave][r + half * 8][lm]      = s[0][r];
      lds_s[wave][r + half * 8][16 + lm] = s[1][r];
    }
    const float* prow = &lds_s[wave][lm][0];
    v4f x0 = *(const v4f*)(prow + half * 8);
    v4f x1 = *(const v4f*)(prow + half * 8 + 4);
    v4f x2 = *(const v4f*)(prow + 16 + half * 8);
    v4f x3 = *(const v4f*)(prow + 16 + half * 8 + 4);
    float sv[16];
#pragma unroll
    for (int i = 0; i < 4; ++i) {
      sv[i] = x0[i]; sv[4 + i] = x1[i]; sv[8 + i] = x2[i]; sv[12 + i] = x3[i];
    }

    // ---- per-lane softmax: row max / exp / row sum ------------------------
    float vmax = sv[0];
#pragma unroll
    for (int i = 1; i < 16; ++i) vmax = fmaxf(vmax, sv[i]);
    vmax = fmaxf(vmax, __shfl_xor(vmax, 16, 32));   // join the two half-lanes
    const float mn = fmaxf(m_l, vmax);
    const float alpha = __expf(m_l - mn);
    m_l = mn;

    v16bf pa;                                       // P in A-fragment layout
    float rs = 0.f;
#pragma unroll
    for (int i = 0; i < 16; ++i) {
      const float p = __expf(sv[i] - mn);
      rs += p;
      pa[i] = (__bf16)p;
    }
    rs += __shfl_xor(rs, 16, 32);
    l_l = l_l * alpha + rs;

    // ---- broadcast alpha back to C layout through a 16-float LDS line -----
    lds_a[wave][lm] = alpha;        // both half-lanes write identical value
    v4f af0 = *(const v4f*)&lds_a[wave][half * 8];
    v4f af1 = *(const v4f*)&lds_a[wave][half * 8 + 4];
#pragma unroll
    for (int t = 0; t < 4; ++t) {
#pragma unroll
      for (int r = 0; r < 4; ++r) {
        o[t][r]     *= af0[r];
        o[t][4 + r] *= af1[r];
      }
    }

    // ---- O += P(16x32) . V(32x64); Vt layout makes B frags contiguous -----
#pragma unroll
    for (int t = 0; t < 4; ++t) {
      const __bf16* vcol =
          Vt + ((size_t)bh * DH + t * 16 + lm) * TSEQ + kt + half * 8;
      const v16bf vb = afrag_bf16(vcol);
      o[t] = __builtin_amdgcn_wmma_f32_16x16x32_bf16(
          false, pa, false, vb, (short)0, o[t], false, false);
    }
  }

  // ---- epilogue: O /= l (broadcast through LDS), write f32 [B*T, EMB] ------
  lds_a[wave][lm] = 1.f / l_l;
  v4f iv0 = *(const v4f*)&lds_a[wave][half * 8];
  v4f iv1 = *(const v4f*)&lds_a[wave][half * 8 + 4];

  const int b = bh >> 4, h = bh & 15;
#pragma unroll
  for (int r = 0; r < 8; ++r) {
    const float inv = (r < 4) ? iv0[r & 3] : iv1[r & 3];
    const int m = b * TSEQ + q0 + r + half * 8;
#pragma unroll
    for (int t = 0; t < 4; ++t) {
      const int n = h * DH + t * 16 + lm;
      Attn[(size_t)m * EMB + n] = o[t][r] * inv;
    }
  }
}

// ---------------------------------------------------------------------------
extern "C" void kernel_launch(void* const* d_in, const int* in_sizes, int n_in,
                              void* d_out, int out_size, void* d_ws, size_t ws_size,
                              hipStream_t stream) {
  const float* query = (const float*)d_in[0];
  const float* key   = (const float*)d_in[1];
  const float* value = (const float*)d_in[2];
  const float* Wq    = (const float*)d_in[3];
  const float* bq    = (const float*)d_in[4];
  const float* Wk    = (const float*)d_in[5];
  const float* bk    = (const float*)d_in[6];
  const float* Wv    = (const float*)d_in[7];
  const float* bv    = (const float*)d_in[8];
  const float* Wo    = (const float*)d_in[9];
  const float* bo    = (const float*)d_in[10];

  const size_t nQKV = (size_t)MROWS * EMB;        // 4096*1024
  char* ws = (char*)d_ws;
  __bf16* Qb  = (__bf16*)(ws);                    //  8 MiB
  __bf16* Kb  = (__bf16*)(ws + nQKV * 2);         //  8 MiB
  __bf16* Vt  = (__bf16*)(ws + nQKV * 4);         //  8 MiB
  float*  Att = (float*) (ws + nQKV * 6);         // 16 MiB

  dim3 gblock(128);
  dim3 ggrid(MROWS / 32, EMB / 256);              // (128, 4)

  // Score scale 1/sqrt(DH) folded into the Q projection epilogue.
  proj_gemm_wmma<0><<<ggrid, gblock, 0, stream>>>(query, Wq, bq, Qb, EMB, 0.125f);
  proj_gemm_wmma<0><<<ggrid, gblock, 0, stream>>>(key,   Wk, bk, Kb, EMB, 1.0f);
  proj_gemm_wmma<1><<<ggrid, gblock, 0, stream>>>(value, Wv, bv, Vt, EMB, 1.0f);

  flash_attn_wmma<<<dim3(TSEQ / 64, BATCH * HEADS), gblock, 0, stream>>>(
      Qb, Kb, Vt, Att);

  proj_gemm_wmma<2><<<ggrid, gblock, 0, stream>>>(Att, Wo, bo, d_out, EMB, 1.0f);
}
